// DualTransformerLayer_23416161697999
// MI455X (gfx1250) — matrix operand order repaired and checked by reading, not compiled, not run on hardware
//
#include <hip/hip_runtime.h>
#include <hip/hip_bf16.h>
#include <math.h>

// ---------------------------------------------------------------------------
// Problem constants (from reference)
// ---------------------------------------------------------------------------
constexpr int BATCH  = 32;
constexpr int SEQ    = 379;
constexpr int HDIM   = 512;
constexpr int NHEADS = 8;
constexpr int DK     = 64;              // HDIM / NHEADS
constexpr int P0     = 4;
constexpr int PR     = 5;
constexpr int HID    = 256;
constexpr long long BN  = (long long)BATCH * SEQ;          // 12128
constexpr long long NN2 = (long long)SEQ * SEQ;            // 143641
constexpr float SCALE_QK = 0.125f;                          // 64^-0.5
// TAU = 2.0, WF = WS = 0.5

// padded score-matrix row pitch: 16B-aligned rows so the ctx GEMM (probs@v)
// can stage A tiles with b128 vector loads
constexpr int SPITCH = 384;
constexpr long long SROW = (long long)SEQ * SPITCH;        // per-(b,h) stride

// ---------------------------------------------------------------------------
// WMMA types (gfx1250, wave32)
// ---------------------------------------------------------------------------
typedef __attribute__((ext_vector_type(16))) __bf16 bf16x16;
typedef __attribute__((ext_vector_type(8)))  __bf16 bf16x8;
typedef __attribute__((ext_vector_type(8)))  float  f32x8;

constexpr int LPAD = 40;   // padded LDS row stride (bf16 elems): 80B -> bank-conflict-free b128 reads

// ---------------------------------------------------------------------------
// Batched mixed-precision GEMM: C = beta*act(A@B + bias) + alpha*res
//   A fp32 (M,K) with row-stride lda; batch offset = b*asB + h*asH
//   B fp32: btrans==0 -> element(k,n) at k*ldb+n ; btrans==1 -> n*ldb+k
//   Block tile 128(M) x 64(N), K-step 32, 256 threads = 8 wave32 waves.
//   Each wave computes a 16x64 strip via 4x v_wmma_f32_16x16x32_bf16.
//   Staging is vectorized (b128 loads + packed bf16 cvt + b128 LDS stores)
//   on the aligned/full-tile fast path; scalar guarded fallback at edges.
//   act: 0=none, 1=exact GELU, 2=ReLU.  res only used when batches==1.
// ---------------------------------------------------------------------------
__global__ __launch_bounds__(256)
void wmma_gemm_kernel(const float* __restrict__ A, long long asB, long long asH, int lda,
                      const float* __restrict__ Bm, long long bsB, long long bsH, int ldb, int btrans,
                      float* __restrict__ C, long long csB, long long csH, int ldc,
                      const float* __restrict__ bias,
                      const float* __restrict__ res, int ldr,
                      int M, int Nout, int K,
                      int heads, int act, float alpha, float beta)
{
    const int bz = blockIdx.z;
    const int bb = bz / heads;
    const int hh = bz % heads;
    A  += (long long)bb * asB + (long long)hh * asH;
    Bm += (long long)bb * bsB + (long long)hh * bsH;
    C  += (long long)bb * csB + (long long)hh * csH;

    // runtime fast-path eligibility: 16B-aligned base + row stride multiple of 4
    const bool a_vec = ((((size_t)A)  & 15) == 0) && ((lda & 3) == 0);
    const bool b_vec = ((((size_t)Bm) & 15) == 0) && ((ldb & 3) == 0);

    __shared__ __bf16 As[128 * LPAD];   // [m][k], padded
    __shared__ __bf16 Bs[64 * LPAD];    // [n][k], padded (K contiguous per out column)

    const int tid  = threadIdx.x;
    const int wave = tid >> 5;
    const int lane = tid & 31;
    const int lr   = lane & 15;       // row/col-in-tile selector
    const int lh   = lane >> 4;       // half selector

    const int row0 = blockIdx.x * 128;
    const int col0 = blockIdx.y * 64;

    f32x8 acc[4];
#pragma unroll
    for (int t = 0; t < 4; ++t)
#pragma unroll
        for (int r = 0; r < 8; ++r) acc[t][r] = 0.0f;

    const int ar = tid >> 1;            // A staging: row           (2 thr/row)
    const int ak = (tid & 1) * 16;      // A staging: k start (16 elems)
    const int bn = tid >> 2;            // B staging (btrans=1): out col
    const int bk = (tid & 3) * 8;       // B staging (btrans=1): k start
    const int bkr = tid >> 3;           // B staging (btrans=0): k row (0..31)
    const int bns = (tid & 7) * 8;      // B staging (btrans=0): n start

    for (int k0 = 0; k0 < K; k0 += 32) {
        const bool kfull = (k0 + 32 <= K);

        // ---- stage A tile (fp32 -> bf16) ----
        {
            const int garow = row0 + ar;
            if (a_vec && kfull && garow < M) {
                const float* ap = A + (long long)garow * lda + k0 + ak;
                if (k0 + 64 <= K) __builtin_prefetch(ap + 32, 0, 1);
                const float4 f0 = *reinterpret_cast<const float4*>(ap + 0);
                const float4 f1 = *reinterpret_cast<const float4*>(ap + 4);
                const float4 f2 = *reinterpret_cast<const float4*>(ap + 8);
                const float4 f3 = *reinterpret_cast<const float4*>(ap + 12);
                bf16x8 p0, p1;
                p0[0] = (__bf16)f0.x; p0[1] = (__bf16)f0.y; p0[2] = (__bf16)f0.z; p0[3] = (__bf16)f0.w;
                p0[4] = (__bf16)f1.x; p0[5] = (__bf16)f1.y; p0[6] = (__bf16)f1.z; p0[7] = (__bf16)f1.w;
                p1[0] = (__bf16)f2.x; p1[1] = (__bf16)f2.y; p1[2] = (__bf16)f2.z; p1[3] = (__bf16)f2.w;
                p1[4] = (__bf16)f3.x; p1[5] = (__bf16)f3.y; p1[6] = (__bf16)f3.z; p1[7] = (__bf16)f3.w;
                *reinterpret_cast<bf16x8*>(&As[ar * LPAD + ak])     = p0;
                *reinterpret_cast<bf16x8*>(&As[ar * LPAD + ak + 8]) = p1;
            } else {
#pragma unroll
                for (int i = 0; i < 16; ++i) {
                    const int kk = ak + i;
                    float v = 0.0f;
                    if (garow < M && (k0 + kk) < K) v = A[(long long)garow * lda + k0 + kk];
                    As[ar * LPAD + kk] = (__bf16)v;
                }
            }
        }
        // ---- stage B tile (fp32 -> bf16, stored [n][k]) ----
        if (btrans) {
            const int gcol = col0 + bn;
            if (b_vec && kfull && gcol < Nout) {
                const float* bp = Bm + (long long)gcol * ldb + k0 + bk;
                const float4 f0 = *reinterpret_cast<const float4*>(bp + 0);
                const float4 f1 = *reinterpret_cast<const float4*>(bp + 4);
                bf16x8 p0;
                p0[0] = (__bf16)f0.x; p0[1] = (__bf16)f0.y; p0[2] = (__bf16)f0.z; p0[3] = (__bf16)f0.w;
                p0[4] = (__bf16)f1.x; p0[5] = (__bf16)f1.y; p0[6] = (__bf16)f1.z; p0[7] = (__bf16)f1.w;
                *reinterpret_cast<bf16x8*>(&Bs[bn * LPAD + bk]) = p0;
            } else {
#pragma unroll
                for (int i = 0; i < 8; ++i) {
                    const int kk = bk + i;
                    const int gk = k0 + kk;
                    float v = 0.0f;
                    if (gcol < Nout && gk < K) v = Bm[(long long)gcol * ldb + gk];
                    Bs[bn * LPAD + kk] = (__bf16)v;
                }
            }
        } else {
            const int gk = k0 + bkr;
            if (b_vec && gk < K && (col0 + bns + 8) <= Nout) {
                const float* bp = Bm + (long long)gk * ldb + col0 + bns;
                const float4 f0 = *reinterpret_cast<const float4*>(bp + 0);
                const float4 f1 = *reinterpret_cast<const float4*>(bp + 4);
                Bs[(bns + 0) * LPAD + bkr] = (__bf16)f0.x;
                Bs[(bns + 1) * LPAD + bkr] = (__bf16)f0.y;
                Bs[(bns + 2) * LPAD + bkr] = (__bf16)f0.z;
                Bs[(bns + 3) * LPAD + bkr] = (__bf16)f0.w;
                Bs[(bns + 4) * LPAD + bkr] = (__bf16)f1.x;
                Bs[(bns + 5) * LPAD + bkr] = (__bf16)f1.y;
                Bs[(bns + 6) * LPAD + bkr] = (__bf16)f1.z;
                Bs[(bns + 7) * LPAD + bkr] = (__bf16)f1.w;
            } else {
#pragma unroll
                for (int i = 0; i < 8; ++i) {
                    const int gcol = col0 + bns + i;
                    float v = 0.0f;
                    if (gcol < Nout && gk < K) v = Bm[(long long)gk * ldb + gcol];
                    Bs[(bns + i) * LPAD + bkr] = (__bf16)v;
                }
            }
        }
        __syncthreads();

        // ---- A fragment: 16x32 bf16 (ISA 7.12.2 layout) ----
        bf16x16 af;
        {
            const int arow = wave * 16 + lr;
#pragma unroll
            for (int i = 0; i < 16; ++i) {
                const int v   = i >> 1;
                const int pos = i & 1;
                const int kk  = 2 * v + pos + 8 * lh + ((v >= 4) ? 8 : 0);
                af[i] = As[arow * LPAD + kk];
            }
        }
        // ---- 4 WMMAs over the 64-wide N strip ----
#pragma unroll
        for (int t = 0; t < 4; ++t) {
            bf16x16 bf;
            const int ncol = t * 16 + lr;
            const int kg   = lh * 8;
#pragma unroll
            for (int i = 0; i < 16; ++i) bf[i] = Bs[ncol * LPAD + kg + i + ((i >= 8) ? 8 : 0)];
            acc[t] = __builtin_amdgcn_wmma_f32_16x16x32_bf16(
                false, af, false, bf, (short)0, acc[t], false, false);
        }
        __syncthreads();
    }

    // ---- epilogue ----
#pragma unroll
    for (int t = 0; t < 4; ++t) {
        const int col = col0 + t * 16 + lr;
        if (col >= Nout) continue;
        const float bv = bias ? bias[col] : 0.0f;
#pragma unroll
        for (int r = 0; r < 8; ++r) {
            const int row = row0 + wave * 16 + lh * 8 + r;
            if (row >= M) continue;
            float v = acc[t][r] + bv;
            if (act == 1)      v = 0.5f * v * (1.0f + erff(v * 0.70710678118654752f));
            else if (act == 2) v = v > 0.0f ? v : 0.0f;
            float o = beta * v;
            if (res) o += alpha * res[(long long)row * ldr + col];
            C[(long long)row * ldc + col] = o;
        }
    }
}

// ---------------------------------------------------------------------------
// LayerNorm over rows of length 512 (one block of 256 threads per row)
// ---------------------------------------------------------------------------
__global__ __launch_bounds__(256)
void layernorm_kernel(const float* __restrict__ X, const float* __restrict__ g,
                      const float* __restrict__ b, float* __restrict__ Y)
{
    const long long row = blockIdx.x;
    const float* x = X + row * HDIM;
    float*       y = Y + row * HDIM;
    __shared__ float red[256];
    const int tid = threadIdx.x;
    const float x0 = x[tid], x1 = x[tid + 256];
    red[tid] = x0 + x1;
    __syncthreads();
    for (int s = 128; s > 0; s >>= 1) { if (tid < s) red[tid] += red[tid + s]; __syncthreads(); }
    const float mu = red[0] * (1.0f / HDIM);
    __syncthreads();
    const float d0 = x0 - mu, d1 = x1 - mu;
    red[tid] = d0 * d0 + d1 * d1;
    __syncthreads();
    for (int s = 128; s > 0; s >>= 1) { if (tid < s) red[tid] += red[tid + s]; __syncthreads(); }
    const float rs = rsqrtf(red[0] * (1.0f / HDIM) + 1e-5f);
    y[tid]       = d0 * rs * g[tid]       + b[tid];
    y[tid + 256] = d1 * rs * g[tid + 256] + b[tid + 256];
}

// log_softmax over rows of length 512
__global__ __launch_bounds__(256)
void logsoftmax_kernel(const float* __restrict__ X, float* __restrict__ L)
{
    const long long row = blockIdx.x;
    const float* x = X + row * HDIM;
    float*       l = L + row * HDIM;
    __shared__ float red[256];
    const int tid = threadIdx.x;
    const float x0 = x[tid], x1 = x[tid + 256];
    red[tid] = fmaxf(x0, x1);
    __syncthreads();
    for (int s = 128; s > 0; s >>= 1) { if (tid < s) red[tid] = fmaxf(red[tid], red[tid + s]); __syncthreads(); }
    const float mx = red[0];
    __syncthreads();
    red[tid] = expf(x0 - mx) + expf(x1 - mx);
    __syncthreads();
    for (int s = 128; s > 0; s >>= 1) { if (tid < s) red[tid] += red[tid + s]; __syncthreads(); }
    const float lse = mx + logf(red[0]);
    l[tid] = x0 - lse;
    l[tid + 256] = x1 - lse;
}

// zbar[b][c] = mean_n z[b][n][c]
__global__ void rowmean_kernel(const float* __restrict__ Z, float* __restrict__ zbar, int total)
{
    const int idx = blockIdx.x * blockDim.x + threadIdx.x;
    if (idx >= total) return;
    const int b = idx / HDIM, c = idx % HDIM;
    const float* p = Z + (long long)b * SEQ * HDIM + c;
    float s = 0.0f;
    for (int t = 0; t < SEQ; ++t) s += p[(long long)t * HDIM];
    zbar[idx] = s * (1.0f / SEQ);
}

// gated softmax over padded score rows: row = softmax(s*SCALE*(1+G))
__global__ __launch_bounds__(128)
void softmax_gate_kernel(float* __restrict__ scores, const float* __restrict__ G)
{
    const long long rid = blockIdx.x;                 // (b*H + h)*SEQ + i
    const int i  = (int)(rid % SEQ);
    const long long bh = rid / SEQ;
    const int b  = (int)(bh / NHEADS);
    float*       row  = scores + bh * SROW + (long long)i * SPITCH;
    const float* grow = G + ((long long)b * SEQ + i) * SEQ;
    __shared__ float red[128];
    const int tid = threadIdx.x;
    float vals[4];
    float mx = -3.0e38f;
    int cnt = 0;
    for (int j = tid; j < SEQ; j += 128) {
        const float v = row[j] * SCALE_QK * (1.0f + grow[j]);
        vals[cnt++] = v;
        mx = fmaxf(mx, v);
    }
    red[tid] = mx; __syncthreads();
    for (int s = 64; s > 0; s >>= 1) { if (tid < s) red[tid] = fmaxf(red[tid], red[tid + s]); __syncthreads(); }
    mx = red[0]; __syncthreads();
    float sum = 0.0f;
    for (int c = 0; c < cnt; ++c) { vals[c] = expf(vals[c] - mx); sum += vals[c]; }
    red[tid] = sum; __syncthreads();
    for (int s = 64; s > 0; s >>= 1) { if (tid < s) red[tid] += red[tid + s]; __syncthreads(); }
    const float inv = 1.0f / red[0];
    cnt = 0;
    for (int j = tid; j < SEQ; j += 128) row[j] = vals[cnt++] * inv;
}

// out = a + b
__global__ void add_kernel(const float* __restrict__ a, const float* __restrict__ b,
                           float* __restrict__ o, long long total)
{
    for (long long id = (long long)blockIdx.x * blockDim.x + threadIdx.x; id < total;
         id += (long long)gridDim.x * blockDim.x)
        o[id] = a[id] + b[id];
}

// alpha = tanh(a2out) * (1 + scale_r[p])
__global__ void alpha_kernel(float* __restrict__ a, const float* __restrict__ scale_r, int total)
{
    const int id = blockIdx.x * blockDim.x + threadIdx.x;
    if (id < total) a[id] = tanhf(a[id]) * (1.0f + scale_r[id % PR]);
}

// normalize rows of length SEQ: u /= max(||u||, 1e-12)
__global__ __launch_bounds__(128)
void normalize_kernel(float* __restrict__ U)
{
    float* u = U + (long long)blockIdx.x * SEQ;
    __shared__ float red[128];
    const int tid = threadIdx.x;
    float s = 0.0f;
    for (int j = tid; j < SEQ; j += 128) { const float v = u[j]; s += v * v; }
    red[tid] = s; __syncthreads();
    for (int t = 64; t > 0; t >>= 1) { if (tid < t) red[tid] += red[tid + t]; __syncthreads(); }
    const float inv = 1.0f / fmaxf(sqrtf(red[0]), 1e-12f);
    for (int j = tid; j < SEQ; j += 128) u[j] *= inv;
}

// Wg_eff = tanh(sym_zero_diag(Wg)) * (1 + scale_g[p])
__global__ void wgeff_kernel(const float* __restrict__ Wg, const float* __restrict__ scale_g,
                             float* __restrict__ out, long long total)
{
    for (long long id = (long long)blockIdx.x * blockDim.x + threadIdx.x; id < total;
         id += (long long)gridDim.x * blockDim.x) {
        const int p = (int)(id / NN2);
        const long long r2 = id % NN2;
        const int i = (int)(r2 / SEQ), j = (int)(r2 % SEQ);
        float w = 0.0f;
        if (i != j)
            w = 0.5f * (Wg[(long long)p * NN2 + (long long)i * SEQ + j] +
                        Wg[(long long)p * NN2 + (long long)j * SEQ + i]);
        out[id] = tanhf(w) * (1.0f + scale_g[p]);
    }
}

// T[b,i,j] = sum_p (Wg_eff[p,i,j] + Wr[b,p,i,j]) * P_bin[b,p,i,j]
__global__ void apf_T_kernel(const float* __restrict__ pri, const float* __restrict__ WgE,
                             const float* __restrict__ UA, const float* __restrict__ UB,
                             const float* __restrict__ alpha, float* __restrict__ T)
{
    const long long total = (long long)BATCH * NN2;
    for (long long id = (long long)blockIdx.x * blockDim.x + threadIdx.x; id < total;
         id += (long long)gridDim.x * blockDim.x) {
        const int j = (int)(id % SEQ);
        const long long t = id / SEQ;
        const int i = (int)(t % SEQ);
        const int b = (int)(t / SEQ);
        float mx = 0.0f, s = 0.0f;
#pragma unroll
        for (int p = 0; p < PR; ++p) {
            float pb;
            if (p < P0) {
                const float pv = pri[(((long long)b * P0 + p) * SEQ + i) * SEQ + j];
                mx = fmaxf(mx, pv);
                pb = pv > 0.0f ? 1.0f : 0.0f;
            } else {
                pb = (1.0f - mx) > 0.0f ? 1.0f : 0.0f;   // free prior
            }
            if (pb != 0.0f) {
                const float a = alpha[b * PR + p];
                const float* ua = UA + ((long long)b * PR + p) * SEQ;
                const float* ub = UB + ((long long)b * PR + p) * SEQ;
                const float wr = (i == j) ? 0.0f
                                          : a * 0.5f * (ua[i] * ub[j] + ub[i] * ua[j]);
                s += WgE[(long long)p * NN2 + (long long)i * SEQ + j] + wr;
            }
        }
        T[id] = s;
    }
}

// G = tanh(clip(0.5*(T + T^T)/TAU, -6, 6)), zero diagonal
__global__ void apf_G_kernel(const float* __restrict__ T, float* __restrict__ G)
{
    const long long total = (long long)BATCH * NN2;
    for (long long id = (long long)blockIdx.x * blockDim.x + threadIdx.x; id < total;
         id += (long long)gridDim.x * blockDim.x) {
        const int b = (int)(id / NN2);
        const long long r = id % NN2;
        const int i = (int)(r / SEQ), j = (int)(r % SEQ);
        float v = 0.0f;
        if (i != j) {
            v = 0.25f * (T[(long long)b * NN2 + (long long)i * SEQ + j] +
                         T[(long long)b * NN2 + (long long)j * SEQ + i]);   // *0.5 (sym), /TAU=2
            v = tanhf(fminf(6.0f, fmaxf(-6.0f, v)));
        }
        G[id] = v;
    }
}

__global__ void set_zero_kernel(float* p) { if (threadIdx.x == 0 && blockIdx.x == 0) *p = 0.0f; }

// loss += 0.5*(WF*KL(l_FC,p_SC) + WS*KL(l_SC,p_FC)), batchmean over BN rows
__global__ __launch_bounds__(256)
void kl_kernel(const float* __restrict__ Lf, const float* __restrict__ Ls,
               float* __restrict__ loss, long long total, float invrows)
{
    __shared__ float red[256];
    const int tid = threadIdx.x;
    float s = 0.0f;
    for (long long id = (long long)blockIdx.x * blockDim.x + tid; id < total;
         id += (long long)gridDim.x * blockDim.x) {
        const float lf = Lf[id], ls = Ls[id];
        s += 0.5f * (0.5f * expf(ls) * (ls - lf) + 0.5f * expf(lf) * (lf - ls));
    }
    red[tid] = s; __syncthreads();
    for (int t = 128; t > 0; t >>= 1) { if (tid < t) red[tid] += red[tid + t]; __syncthreads(); }
    if (tid == 0) atomicAdd(loss, red[0] * invrows);
}

// ---------------------------------------------------------------------------
// Host-side orchestration
// ---------------------------------------------------------------------------
struct BP {
    const float *lnA_g, *lnA_b, *lnF_g, *lnF_b, *qkv_w;
    const float *ffn1_w, *ffn1_b, *ffn2_w, *ffn2_b;
    const float *Wg, *uA1_w, *uA1_b, *uA2_w, *uA2_b, *uB1_w, *uB1_b, *uB2_w, *uB2_b;
    const float *a1_w, *a1_b, *a2_w, *a2_b, *scale_g, *scale_r;
    const float *din_w, *din_b, *dout_w, *dout_b;
};

static inline void gemm(hipStream_t st,
    const float* A, long long asB, long long asH, int lda,
    const float* Bm, long long bsB, long long bsH, int ldb, int btrans,
    float* C, long long csB, long long csH, int ldc,
    const float* bias, const float* res, int ldr,
    int M, int Nout, int K, int batches, int heads,
    int act, float alpha, float beta)
{
    dim3 grid((M + 127) / 128, (Nout + 63) / 64, batches);
    wmma_gemm_kernel<<<grid, dim3(256), 0, st>>>(A, asB, asH, lda, Bm, bsB, bsH, ldb, btrans,
        C, csB, csH, ldc, bias, res, ldr, M, Nout, K, heads, act, alpha, beta);
}

extern "C" void kernel_launch(void* const* d_in, const int* in_sizes, int n_in,
                              void* d_out, int out_size, void* d_ws, size_t ws_size,
                              hipStream_t stream)
{
    (void)in_sizes; (void)n_in; (void)out_size; (void)ws_size;
    auto F = [&](int i) { return (const float*)d_in[i]; };

    // setup_inputs order: x_FC, x_SC, pri_FC, pri_SC, FC, SC, params{FC..., SC...}
    const float* x_FC   = F(0);
    const float* x_SC   = F(1);
    const float* pri_FC = F(2);
    const float* pri_SC = F(3);
    // d_in[4], d_in[5] = FC, SC (unused in mask mode)

    auto mk = [&](int o) {
        BP p;
        p.lnA_g = F(o + 0);  p.lnA_b = F(o + 1);  p.lnF_g = F(o + 2);  p.lnF_b = F(o + 3);
        p.qkv_w = F(o + 4);
        p.ffn1_w = F(o + 5); p.ffn1_b = F(o + 6); p.ffn2_w = F(o + 7); p.ffn2_b = F(o + 8);
        p.Wg = F(o + 9);
        p.uA1_w = F(o + 10); p.uA1_b = F(o + 11); p.uA2_w = F(o + 12); p.uA2_b = F(o + 13);
        p.uB1_w = F(o + 14); p.uB1_b = F(o + 15); p.uB2_w = F(o + 16); p.uB2_b = F(o + 17);
        p.a1_w = F(o + 18);  p.a1_b = F(o + 19);  p.a2_w = F(o + 20);  p.a2_b = F(o + 21);
        p.scale_g = F(o + 22); p.scale_r = F(o + 23);
        p.din_w = F(o + 24); p.din_b = F(o + 25); p.dout_w = F(o + 26); p.dout_b = F(o + 27);
        return p;
    };
    const BP pFC = mk(6), pSC = mk(34);

    // Output layout: xf_new | xs_new | loss | G_FC | G_SC
    float* outp = (float*)d_out;
    float* xf_out  = outp;
    float* xs_out  = outp + BN * HDIM;
    float* loss_p  = outp + 2 * BN * HDIM;
    float* Gfc_out = loss_p + 1;
    float* Gsc_out = Gfc_out + (long long)BATCH * NN2;

    // Workspace carve-up
    float* ws = (float*)d_ws;
    size_t off = 0;
    auto wsalloc = [&](size_t n) { float* p = ws + off; off += n; return p; };
    float* z      = wsalloc((size_t)BN * HDIM);
    float* z2     = wsalloc((size_t)BN * HDIM);
    float* qkv    = wsalloc((size_t)BN * 3 * HDIM);
    float* scores = wsalloc((size_t)BATCH * NHEADS * SROW);   // padded pitch
    float* ctx    = wsalloc((size_t)BN * HDIM);
    float* xnew   = wsalloc((size_t)BN * HDIM);
    float* h1     = wsalloc((size_t)BN * 4 * HDIM);
    float* proj   = wsalloc((size_t)BN * HDIM);
    float* lFC    = wsalloc((size_t)BN * HDIM);
    float* lSC    = wsalloc((size_t)BN * HDIM);
    float* zbar   = wsalloc((size_t)BATCH * HDIM);
    float* mlph   = wsalloc((size_t)BATCH * HID);
    float* UA     = wsalloc((size_t)BATCH * PR * SEQ);
    float* UB     = wsalloc((size_t)BATCH * PR * SEQ);
    float* alp    = wsalloc((size_t)BATCH * PR);
    float* WgE    = wsalloc((size_t)PR * NN2);
    float* Tbuf   = wsalloc((size_t)BATCH * NN2);

    const int H3 = 3 * HDIM;

    auto run_branch = [&](const BP& p, const float* x, const float* pri,
                          float* Gout, float* Lbuf, float* xout) {
        // z = LN_attn(x)
        layernorm_kernel<<<(int)BN, 256, 0, stream>>>(x, p.lnA_g, p.lnA_b, z);
        // zbar = mean_n z
        rowmean_kernel<<<(BATCH * HDIM + 255) / 256, 256, 0, stream>>>(z, zbar, BATCH * HDIM);
        // qkv = z @ Wqkv
        gemm(stream, z, 0, 0, HDIM, p.qkv_w, 0, 0, H3, 0, qkv, 0, 0, H3,
             nullptr, nullptr, 0, (int)BN, H3, HDIM, 1, 1, 0, 0.0f, 1.0f);

        // ---- APF mask ----
        gemm(stream, zbar, 0, 0, HDIM, p.uA1_w, 0, 0, HID, 0, mlph, 0, 0, HID,
             p.uA1_b, nullptr, 0, BATCH, HID, HDIM, 1, 1, 2, 0.0f, 1.0f);
        gemm(stream, mlph, 0, 0, HID, p.uA2_w, 0, 0, PR * SEQ, 0, UA, 0, 0, PR * SEQ,
             p.uA2_b, nullptr, 0, BATCH, PR * SEQ, HID, 1, 1, 0, 0.0f, 1.0f);
        gemm(stream, zbar, 0, 0, HDIM, p.uB1_w, 0, 0, HID, 0, mlph, 0, 0, HID,
             p.uB1_b, nullptr, 0, BATCH, HID, HDIM, 1, 1, 2, 0.0f, 1.0f);
        gemm(stream, mlph, 0, 0, HID, p.uB2_w, 0, 0, PR * SEQ, 0, UB, 0, 0, PR * SEQ,
             p.uB2_b, nullptr, 0, BATCH, PR * SEQ, HID, 1, 1, 0, 0.0f, 1.0f);
        gemm(stream, zbar, 0, 0, HDIM, p.a1_w, 0, 0, HID, 0, mlph, 0, 0, HID,
             p.a1_b, nullptr, 0, BATCH, HID, HDIM, 1, 1, 2, 0.0f, 1.0f);
        gemm(stream, mlph, 0, 0, HID, p.a2_w, 0, 0, PR, 0, alp, 0, 0, PR,
             p.a2_b, nullptr, 0, BATCH, PR, HID, 1, 1, 0, 0.0f, 1.0f);
        alpha_kernel<<<1, 256, 0, stream>>>(alp, p.scale_r, BATCH * PR);
        normalize_kernel<<<BATCH * PR, 128, 0, stream>>>(UA);
        normalize_kernel<<<BATCH * PR, 128, 0, stream>>>(UB);
        wgeff_kernel<<<1024, 256, 0, stream>>>(p.Wg, p.scale_g, WgE, (long long)PR * NN2);
        apf_T_kernel<<<4096, 256, 0, stream>>>(pri, WgE, UA, UB, alp, Tbuf);
        apf_G_kernel<<<4096, 256, 0, stream>>>(Tbuf, Gout);

        // ---- attention ----
        // scores[b,h,i,j] = q . k  (q at col 0, k at col HDIM of packed qkv)
        gemm(stream,
             qkv,            (long long)SEQ * H3, DK, H3,
             qkv + HDIM,     (long long)SEQ * H3, DK, H3, 1,
             scores,         (long long)NHEADS * SROW, SROW, SPITCH,
             nullptr, nullptr, 0,
             SEQ, SEQ, DK, BATCH * NHEADS, NHEADS, 0, 0.0f, 1.0f);
        softmax_gate_kernel<<<(int)(BATCH * NHEADS * SEQ), 128, 0, stream>>>(scores, Gout);
        // ctx[b,i,h*64+d] = probs @ v  (v at col 2*HDIM); padded-pitch A enables b128 staging
        gemm(stream,
             scores,         (long long)NHEADS * SROW, SROW, SPITCH,
             qkv + 2 * HDIM, (long long)SEQ * H3, DK, H3, 0,
             ctx,            (long long)SEQ * HDIM, DK, HDIM,
             nullptr, nullptr, 0,
             SEQ, DK, SEQ, BATCH * NHEADS, NHEADS, 0, 0.0f, 1.0f);
        add_kernel<<<4096, 256, 0, stream>>>(x, ctx, xnew, BN * HDIM);

        // ---- FFN ----
        layernorm_kernel<<<(int)BN, 256, 0, stream>>>(xnew, p.lnF_g, p.lnF_b, z2);
        gemm(stream, z2, 0, 0, HDIM, p.ffn1_w, 0, 0, 4 * HDIM, 0, h1, 0, 0, 4 * HDIM,
             p.ffn1_b, nullptr, 0, (int)BN, 4 * HDIM, HDIM, 1, 1, 1, 0.0f, 1.0f);
        gemm(stream, h1, 0, 0, 4 * HDIM, p.ffn2_w, 0, 0, HDIM, 0, xnew, 0, 0, HDIM,
             p.ffn2_b, xnew, HDIM, (int)BN, HDIM, 4 * HDIM, 1, 1, 0, 1.0f, 1.0f);

        // ---- distillation head ----
        gemm(stream, xnew, 0, 0, HDIM, p.din_w, 0, 0, HDIM, 0, proj, 0, 0, HDIM,
             p.din_b, nullptr, 0, (int)BN, HDIM, HDIM, 1, 1, 0, 0.0f, 1.0f);
        logsoftmax_kernel<<<(int)BN, 256, 0, stream>>>(proj, Lbuf);
        // x_out = 0.5*xnew + 0.5*(proj @ Wout + b)
        gemm(stream, proj, 0, 0, HDIM, p.dout_w, 0, 0, HDIM, 0, xout, 0, 0, HDIM,
             p.dout_b, xnew, HDIM, (int)BN, HDIM, HDIM, 1, 1, 0, 0.5f, 0.5f);
    };

    run_branch(pFC, x_FC, pri_FC, Gfc_out, lFC, xf_out);
    run_branch(pSC, x_SC, pri_SC, Gsc_out, lSC, xs_out);

    set_zero_kernel<<<1, 32, 0, stream>>>(loss_p);
    kl_kernel<<<1024, 256, 0, stream>>>(lFC, lSC, loss_p, BN * HDIM, 1.0f / (float)BN);
}
